// Vector_Quantizer_14671608283802
// MI455X (gfx1250) — matrix operand order repaired and checked by reading, not compiled, not run on hardware
//
#include <hip/hip_runtime.h>
#include <hip/hip_bf16.h>

typedef __attribute__((ext_vector_type(16))) __bf16 v16bf;
typedef __attribute__((ext_vector_type(8)))  float  v8f;

#define HW      4096            // H*W
#define C_DIM   64
#define K_CODES 1024
#define N_PTS   (32*64*64)      // 131072 points
#define NUMEL   (32*64*64*64)   // 8388608 elements of x / quantized
#define TPW     2               // point tiles per wave (M-blocking)
#define NWAVES  (N_PTS/(16*TPW))// 4096 waves
#define WGS     (NWAVES/8)      // 512 workgroups of 8 waves
#define CCHUNK  128             // codes staged in LDS per chunk
#define NCHUNK  (K_CODES/CCHUNK)

// workspace layout (byte offsets)
#define WS_EHI  0               // 1024*64 bf16  = 128 KB
#define WS_ELO  131072          // 1024*64 bf16  = 128 KB
#define WS_N2   262144          // 1024 f32      =   4 KB
#define WS_PART 266240          // 4096 f32      =  16 KB

#if __has_builtin(__builtin_amdgcn_global_load_async_to_lds_b128)
#define HAS_ASYNC_LDS 1
// builtin signature (from hipcc diagnostic): (v4i AS1*, v4i AS3*, imm int, imm int)
typedef int v4i_t __attribute__((vector_size(4 * sizeof(int))));
typedef __attribute__((address_space(1))) v4i_t* g4p_t;
typedef __attribute__((address_space(3))) v4i_t* l4p_t;
#else
#define HAS_ASYNC_LDS 0
#endif

__device__ __forceinline__ __bf16 f2bf(float f) {
  unsigned u = __builtin_bit_cast(unsigned, f);
  u += 0x7FFFu + ((u >> 16) & 1u);            // round-to-nearest-even
  unsigned short s = (unsigned short)(u >> 16);
  return __builtin_bit_cast(__bf16, s);
}
__device__ __forceinline__ float bf2f(__bf16 b) {
  unsigned u = ((unsigned)__builtin_bit_cast(unsigned short, b)) << 16;
  return __builtin_bit_cast(float, u);
}

// ---- kernel 1: split codebook into bf16 hi/lo, compute ||e||^2 ----
__global__ void vq_prep(const float* __restrict__ emb,
                        __bf16* __restrict__ ehi, __bf16* __restrict__ elo,
                        float* __restrict__ n2) {
  int k = blockIdx.x * blockDim.x + threadIdx.x;
  if (k >= K_CODES) return;
  float acc = 0.f;
  #pragma unroll 4
  for (int c = 0; c < C_DIM; ++c) {
    float v = emb[k * C_DIM + c];
    __bf16 hi = f2bf(v);
    __bf16 lo = f2bf(v - bf2f(hi));
    ehi[k * C_DIM + c] = hi;
    elo[k * C_DIM + c] = lo;
    acc += v * v;
  }
  n2[k] = acc;
}

// ---- kernel 2: WMMA distance sweep + argmax + quantize + loss partials ----
__global__ void __launch_bounds__(256) vq_main(
    const float* __restrict__ x, const float* __restrict__ emb,
    const __bf16* __restrict__ ehi, const __bf16* __restrict__ elo,
    const float* __restrict__ n2, float* __restrict__ outq,
    float* __restrict__ outidx, float* __restrict__ partial) {

  __shared__ __bf16 lds_hi[CCHUNK * C_DIM];   // 16 KB
  __shared__ __bf16 lds_lo[CCHUNK * C_DIM];   // 16 KB
  __shared__ float  lds_n2[CCHUNK];
  __shared__ int    s_k[8][TPW][16];

  const int tid  = threadIdx.x;
  const int lane = tid & 31;
  const int w    = tid >> 5;
  const int gw   = blockIdx.x * 8 + w;        // global wave id
  const int n0   = gw * (16 * TPW);           // first point of this wave
  const int half = lane >> 4;
  const int m    = lane & 15;
  const int nb   = n0 >> 12;                  // HW=4096 block; wave never crosses
  const int off  = n0 & (HW - 1);

  // ---- build A tiles (TPW x 16 points x 64 ch), hi/lo split, contiguous k = 16*half + s ----
  v16bf a_hi[TPW][2], a_lo[TPW][2];
  #pragma unroll
  for (int t = 0; t < TPW; ++t) {
    #pragma unroll
    for (int chunk = 0; chunk < 2; ++chunk) {
      #pragma unroll
      for (int j = 0; j < 8; ++j) {
        int kl = (half << 4) + (j << 1);
        int c0 = chunk * 32 + kl;
        float f0 = x[(nb * C_DIM + c0    ) * HW + off + t * 16 + m];
        float f1 = x[(nb * C_DIM + c0 + 1) * HW + off + t * 16 + m];
        __bf16 h0 = f2bf(f0), h1 = f2bf(f1);
        a_hi[t][chunk][2 * j]     = h0;
        a_hi[t][chunk][2 * j + 1] = h1;
        a_lo[t][chunk][2 * j]     = f2bf(f0 - bf2f(h0));
        a_lo[t][chunk][2 * j + 1] = f2bf(f1 - bf2f(h1));
      }
    }
  }

  float best[TPW][8];
  int   bidx[TPW][8];
  #pragma unroll
  for (int t = 0; t < TPW; ++t)
    #pragma unroll
    for (int i = 0; i < 8; ++i) { best[t][i] = -3.4e38f; bidx[t][i] = 0; }

  // ---- sweep codebook in LDS-staged chunks of CCHUNK codes ----
  for (int ch = 0; ch < NCHUNK; ++ch) {
    __syncthreads();                          // previous chunk fully consumed
    {
      const char* gh = (const char*)(ehi + ch * CCHUNK * C_DIM);
      const char* gl = (const char*)(elo + ch * CCHUNK * C_DIM);
#if HAS_ASYNC_LDS
      #pragma unroll
      for (int i = 0; i < 4; ++i) {           // 256 thr * 16B * 4 = 16 KB each
        int boff = (i * 256 + tid) * 16;
        __builtin_amdgcn_global_load_async_to_lds_b128(
            (g4p_t)(gh + boff), (l4p_t)((char*)lds_hi + boff), 0, 0);
        __builtin_amdgcn_global_load_async_to_lds_b128(
            (g4p_t)(gl + boff), (l4p_t)((char*)lds_lo + boff), 0, 0);
      }
      if (tid < CCHUNK) lds_n2[tid] = n2[ch * CCHUNK + tid];
#if __has_builtin(__builtin_amdgcn_s_wait_asynccnt)
      __builtin_amdgcn_s_wait_asynccnt(0);
#else
      asm volatile("s_wait_asynccnt 0x0" ::: "memory");
#endif
#else
      #pragma unroll
      for (int i = 0; i < 4; ++i) {
        int boff = (i * 256 + tid) * 16;
        *(uint4*)((char*)lds_hi + boff) = *(const uint4*)(gh + boff);
        *(uint4*)((char*)lds_lo + boff) = *(const uint4*)(gl + boff);
      }
      if (tid < CCHUNK) lds_n2[tid] = n2[ch * CCHUNK + tid];
#endif
    }
    __syncthreads();                          // chunk visible to all waves

    #pragma unroll 2
    for (int kt = 0; kt < CCHUNK / 16; ++kt) {
      const int cl   = kt * 16 + m;           // this lane's code within chunk
      const int code = ch * CCHUNK + cl;
      const __bf16* bhp = &lds_hi[cl * C_DIM + (half << 4)];
      const __bf16* blp = &lds_lo[cl * C_DIM + (half << 4)];
      v16bf bh0 = *(const v16bf*)(bhp);
      v16bf bh1 = *(const v16bf*)(bhp + 32);
      v16bf bl0 = *(const v16bf*)(blp);
      v16bf bl1 = *(const v16bf*)(blp + 32);
      float n2v = lds_n2[cl];

      #pragma unroll
      for (int t = 0; t < TPW; ++t) {         // B reused across both M tiles
        v8f acc = {};
        acc = __builtin_amdgcn_wmma_f32_16x16x32_bf16(false, a_lo[t][0], false, bh0, (short)0, acc, false, false);
        acc = __builtin_amdgcn_wmma_f32_16x16x32_bf16(false, a_hi[t][0], false, bl0, (short)0, acc, false, false);
        acc = __builtin_amdgcn_wmma_f32_16x16x32_bf16(false, a_hi[t][0], false, bh0, (short)0, acc, false, false);
        acc = __builtin_amdgcn_wmma_f32_16x16x32_bf16(false, a_lo[t][1], false, bh1, (short)0, acc, false, false);
        acc = __builtin_amdgcn_wmma_f32_16x16x32_bf16(false, a_hi[t][1], false, bl1, (short)0, acc, false, false);
        acc = __builtin_amdgcn_wmma_f32_16x16x32_bf16(false, a_hi[t][1], false, bh1, (short)0, acc, false, false);
        #pragma unroll
        for (int i = 0; i < 8; ++i) {
          float s = 2.0f * acc[i] - n2v;      // argmax == argmin of distance
          if (s > best[t][i]) { best[t][i] = s; bidx[t][i] = code; }
        }
      }
    }
  }

  // ---- per-point argmax across the 16 lanes of each half: shuffle butterfly ----
  #pragma unroll
  for (int o = 1; o < 16; o <<= 1) {
    #pragma unroll
    for (int t = 0; t < TPW; ++t) {
      #pragma unroll
      for (int i = 0; i < 8; ++i) {
        float ov = __shfl_xor(best[t][i], o, 32);
        int   oi = __shfl_xor(bidx[t][i], o, 32);
        if (ov > best[t][i] || (ov == best[t][i] && oi < bidx[t][i])) {
          best[t][i] = ov; bidx[t][i] = oi;
        }
      }
    }
  }
  if (m == 0) {                               // lanes 0 and 16: publish k* (M = i + 8*half)
    #pragma unroll
    for (int t = 0; t < TPW; ++t)
      #pragma unroll
      for (int i = 0; i < 8; ++i) s_k[w][t][half * 8 + i] = bidx[t][i];
  }
  __syncthreads();

  // ---- indices, straight-through quantized, loss partial ----
  if (lane < 16) {
    #pragma unroll
    for (int t = 0; t < TPW; ++t)
      outidx[n0 + t * 16 + lane] = (float)s_k[w][t][lane];
  }
  float lsum = 0.f;
  #pragma unroll
  for (int t = 0; t < TPW; ++t) {
    const int kstar = s_k[w][t][m];
    #pragma unroll 4
    for (int i = 0; i < 32; ++i) {
      int c    = 2 * i + half;                // lanes 0-15: even ch, 16-31: odd ch
      float q  = emb[kstar * C_DIM + c];      // full-precision gather (L2-resident)
      int   xi = (nb * C_DIM + c) * HW + off + t * 16 + m;
      float xv = x[xi];
      outq[xi] = xv + (q - xv);               // mimic reference fp ordering
      float d  = q - xv;
      lsum += d * d;
    }
  }
  #pragma unroll
  for (int o = 16; o > 0; o >>= 1) lsum += __shfl_xor(lsum, o, 32);
  if (lane == 0) partial[gw] = lsum;
}

// ---- kernel 3: deterministic fixed-tree loss reduction ----
__global__ void vq_loss(const float* __restrict__ partial, float* __restrict__ out) {
  __shared__ float red[256];
  int t = threadIdx.x;
  float s = 0.f;
  for (int j = 0; j < NWAVES / 256; ++j) s += partial[t + 256 * j];
  red[t] = s;
  __syncthreads();
  for (int o = 128; o > 0; o >>= 1) {
    if (t < o) red[t] += red[t + o];
    __syncthreads();
  }
  if (t == 0) out[0] = red[0] * (0.25f / (float)NUMEL);
}

extern "C" void kernel_launch(void* const* d_in, const int* in_sizes, int n_in,
                              void* d_out, int out_size, void* d_ws, size_t ws_size,
                              hipStream_t stream) {
  (void)in_sizes; (void)n_in; (void)out_size; (void)ws_size;
  const float* x   = (const float*)d_in[0];
  const float* emb = (const float*)d_in[1];
  float* out = (float*)d_out;
  char*  ws  = (char*)d_ws;
  __bf16* ehi = (__bf16*)(ws + WS_EHI);
  __bf16* elo = (__bf16*)(ws + WS_ELO);
  float*  n2  = (float*)(ws + WS_N2);
  float*  prt = (float*)(ws + WS_PART);

  vq_prep<<<(K_CODES + 255) / 256, 256, 0, stream>>>(emb, ehi, elo, n2);
  vq_main<<<WGS, 256, 0, stream>>>(x, emb, ehi, elo, n2,
                                   out + 1, out + 1 + NUMEL, prt);
  vq_loss<<<1, 256, 0, stream>>>(prt, out);
}
